// Brain_26525718020137
// MI455X (gfx1250) — matrix operand order repaired and checked by reading, not compiled, not run on hardware
//
#include <hip/hip_runtime.h>

// ---------------------------------------------------------------------------
// CDNA5 (gfx1250) fused Mamba-policy forward.
// L==1 collapses the scan: y = (softplus(dt@dtw^T+b)*dot(Bm,Cm) + Dskip)
//                               * silu(u_raw*conv_w[:,3]+conv_b) * silu(z)
// Pipeline: cvt->f16, GEMM1 (WMMA), GEMM2 (WMMA), middle (VALU), GEMM3 (WMMA),
// GEMM4 (WMMA + tanh/clip epilogue).
// GEMMs: 64x128 block tile, 8 waves (2x4), 32x32 per wave (4 WMMA/K-step),
// 3-stage LDS pipeline fed by global_load_async_to_lds_b128; steady-state wait
// is s_wait_asynccnt 3 (newest batch in flight) -> ~2 K-steps latency hiding.
// ---------------------------------------------------------------------------

typedef __attribute__((ext_vector_type(16))) _Float16 v16h;
typedef __attribute__((ext_vector_type(8)))  float    v8f;
typedef __attribute__((ext_vector_type(4)))  int      v4i;

#define BATCH_   8192
#define P_       512
#define DMODEL_  256
#define DINNER_  512

#if __has_builtin(__builtin_amdgcn_global_load_async_to_lds_b128)
#define HAS_ASYNC_LDS 1
#endif

__device__ __forceinline__ void async_b128(const void* g, void* l) {
#ifdef HAS_ASYNC_LDS
  __builtin_amdgcn_global_load_async_to_lds_b128(
      (__attribute__((address_space(1))) v4i*)g,
      (__attribute__((address_space(3))) v4i*)l, 0, 0);
#else
  *(uint4*)l = *(const uint4*)g;
#endif
}

template <int N>
__device__ __forceinline__ void async_wait() {
#ifdef HAS_ASYNC_LDS
#if __has_builtin(__builtin_amdgcn_s_wait_asynccnt)
  __builtin_amdgcn_s_wait_asynccnt(N);
#else
  asm volatile("s_wait_asynccnt %0" ::"i"(N) : "memory");
#endif
#endif
}

// ---------------------------------------------------------------------------
// f32 -> f16 conversion (4-wide, grid-stride)
// ---------------------------------------------------------------------------
__global__ void cvt_f32_f16(const float* __restrict__ src,
                            _Float16* __restrict__ dst, int n4) {
  int i = blockIdx.x * blockDim.x + threadIdx.x;
  int stride = gridDim.x * blockDim.x;
  for (; i < n4; i += stride) {
    float4 v = ((const float4*)src)[i];
    union { _Float16 h[4]; uint2 u; } o;
    o.h[0] = (_Float16)v.x; o.h[1] = (_Float16)v.y;
    o.h[2] = (_Float16)v.z; o.h[3] = (_Float16)v.w;
    ((uint2*)dst)[i] = o.u;
  }
}

// ---------------------------------------------------------------------------
// WMMA GEMM: C[M,N] = A[M,K] * W[N,K]^T  (A,W f16 row-major, f32 accumulate)
// Block: 256 thr = 8 waves (2Mx4N), block tile 64x128, BK=32.
// Wave computes 32x32 via 4 accumulators; LDS tiles row-major over K so each
// lane's fragment is contiguous 16B chunks (ds_load_b128) matching the ISA's
// 16-bit A(16x32)/B(32x16) layouts.  3-stage async global->LDS pipeline.
// EPI: 0 = f16 store (+opt bias0), 1 = f32 store, 2 = mu/log_std epilogue.
// ---------------------------------------------------------------------------
template <int EPI>
__global__ __launch_bounds__(256)
void gemm_wmma(const _Float16* __restrict__ A, const _Float16* __restrict__ W,
               void* __restrict__ Cout, const float* __restrict__ bias0,
               const float* __restrict__ bias1, int M, int N, int K) {
  // per stage: A 64x32 f16 = 4KB at +0, W 128x32 f16 = 8KB at +4096
  __shared__ uint4 lds4[2304];   // 36KB total (3 stages x 12KB)
  char* base = (char*)lds4;

  const int tid  = threadIdx.x;
  const int wave = tid >> 5, lane = tid & 31;
  const int wm = wave & 1, wn = wave >> 1;   // 2(M) x 4(N) wave grid
  const int r  = lane & 15, hi = lane >> 4;
  const int m0 = blockIdx.y * 64;
  const int n0 = blockIdx.x * 128;

  // global->LDS copy mapping (per thread)
  const int arow = tid >> 2, aseg = tid & 3;          // A: 64 rows x 4 segs x 16B
  const int wrow = tid >> 1, wseg = (tid & 1) * 2;    // W: 128 rows x 2x(2 segs)x16B

  const _Float16* Ag = A + (size_t)(m0 + arow) * K + aseg * 8;
  const _Float16* Wg = W + (size_t)(n0 + wrow) * K + wseg * 8;
  char* const aL = base + arow * 64 + aseg * 16;
  char* const wL = base + 4096 + wrow * 64 + wseg * 16;

  auto prefetch = [&](int st, int k0) {
    const int sb = st * 12288;
    async_b128(Ag + k0, aL + sb);
    async_b128(Wg + k0, wL + sb);
    async_b128(Wg + k0 + 8, wL + sb + 16);
  };

  v8f acc[2][2] = {};

  // prologue: fill stages 0 and 1
  prefetch(0, 0);
  if (32 < K) {
    prefetch(1, 32);
    async_wait<3>();   // stage 0 landed (stage 1 may still be in flight)
  } else {
    async_wait<0>();
  }
  __syncthreads();

  int stage = 0;
  for (int k0 = 0; k0 < K; k0 += 32) {
    const bool pf = (k0 + 64 < K);
    if (pf) {
      int s2 = stage + 2; if (s2 >= 3) s2 -= 3;
      prefetch(s2, k0 + 64);
    }

    char* ab = base + stage * 12288;
    char* wb = ab + 4096;
    union { v16h v; uint4 q[2]; } fa[2], fb[2];
#pragma unroll
    for (int mt = 0; mt < 2; ++mt) {
      const int ao = (wm * 32 + mt * 16 + r) * 64 + hi * 16;
      fa[mt].q[0] = *(const uint4*)(ab + ao);
      fa[mt].q[1] = *(const uint4*)(ab + ao + 32);
    }
#pragma unroll
    for (int nt = 0; nt < 2; ++nt) {
      const int bo = (wn * 32 + nt * 16 + r) * 64 + hi * 16;
      fb[nt].q[0] = *(const uint4*)(wb + bo);
      fb[nt].q[1] = *(const uint4*)(wb + bo + 32);
    }
#pragma unroll
    for (int mt = 0; mt < 2; ++mt)
#pragma unroll
      for (int nt = 0; nt < 2; ++nt)
        acc[mt][nt] = __builtin_amdgcn_wmma_f32_16x16x32_f16(
            false, fa[mt].v, false, fb[nt].v, (short)0, acc[mt][nt], false, false);

    // ensure stage (i+1) has landed before anyone reads it next iteration
    if (pf)                    async_wait<3>();   // newest batch may stay in flight
    else if (k0 + 32 < K)      async_wait<0>();   // tail: drain everything
    __syncthreads();
    stage = (stage + 1 == 3) ? 0 : stage + 1;
  }

#pragma unroll
  for (int mt = 0; mt < 2; ++mt) {
    const int mg0 = m0 + wm * 32 + mt * 16 + hi * 8;   // VGPR j -> row mg0+j
#pragma unroll
    for (int nt = 0; nt < 2; ++nt) {
      const int ng = n0 + wn * 32 + nt * 16 + r;       // output col
      const v8f a = acc[mt][nt];
      if (EPI == 0) {
        const float b = bias0 ? bias0[ng] : 0.f;
        _Float16* C = (_Float16*)Cout;
#pragma unroll
        for (int j = 0; j < 8; ++j)
          C[(size_t)(mg0 + j) * N + ng] = (_Float16)(a[j] + b);
      } else if (EPI == 1) {
        float* C = (float*)Cout;
#pragma unroll
        for (int j = 0; j < 8; ++j)
          C[(size_t)(mg0 + j) * N + ng] = a[j];
      } else {
        // cols 0..63 -> mu = tanh(h+mu_b); cols 64..127 -> clip(h+ls_b,-5,2)
        float* O = (float*)Cout;
        if (ng < 64) {
          const float b = bias0[ng];
#pragma unroll
          for (int j = 0; j < 8; ++j)
            O[(size_t)(mg0 + j) * 64 + ng] = tanhf(a[j] + b);
        } else {
          const int nn = ng - 64;
          const float b = bias1[nn];
          float* O2 = O + (size_t)M * 64;
#pragma unroll
          for (int j = 0; j < 8; ++j) {
            float v = a[j] + b;
            v = fminf(2.0f, fmaxf(-5.0f, v));
            O2[(size_t)(mg0 + j) * 64 + nn] = v;
          }
        }
      }
    }
  }
}

// ---------------------------------------------------------------------------
// Collapsed Mamba middle.  One wave per batch row, 8 rows per block.
// ---------------------------------------------------------------------------
__global__ __launch_bounds__(256)
void mamba_mid(const float* __restrict__ XZ, const float* __restrict__ conv_w,
               const float* __restrict__ conv_b, const float* __restrict__ x_proj_w,
               const float* __restrict__ dt_proj_w, const float* __restrict__ dt_proj_b,
               const float* __restrict__ Dskip, _Float16* __restrict__ Y) {
  __shared__ float u_sh[8][DINNER_];
  __shared__ float xd_sh[8][48];
  const int wave = threadIdx.x >> 5, lane = threadIdx.x & 31;
  const int b = blockIdx.x * 8 + wave;
  const float* xz = XZ + (size_t)b * (2 * DINNER_);

#pragma unroll 4
  for (int t = 0; t < 16; ++t) {
    const int c = t * 32 + lane;
    const float raw = xz[c] * conv_w[c * 4 + 3] + conv_b[c];
    u_sh[wave][c] = raw / (1.f + __expf(-raw));
  }
  __syncthreads();

  {
    const int j = lane;
    const float* wrow = x_proj_w + (size_t)j * DINNER_;
    float a0 = 0.f;
    for (int c = 0; c < DINNER_; ++c) a0 += wrow[c] * u_sh[wave][c];
    xd_sh[wave][j] = a0;
    if (lane < 16) {
      const int j2 = 32 + lane;
      const float* wrow2 = x_proj_w + (size_t)j2 * DINNER_;
      float a2 = 0.f;
      for (int c = 0; c < DINNER_; ++c) a2 += wrow2[c] * u_sh[wave][c];
      xd_sh[wave][j2] = a2;
    }
  }
  __syncthreads();

  float s = 0.f;
#pragma unroll
  for (int n = 0; n < 16; ++n) s += xd_sh[wave][16 + n] * xd_sh[wave][32 + n];

  for (int t = 0; t < 16; ++t) {
    const int c = t * 32 + lane;
    float d = dt_proj_b[c];
    const float* dwr = dt_proj_w + (size_t)c * 16;
#pragma unroll
    for (int rr = 0; rr < 16; ++rr) d += dwr[rr] * xd_sh[wave][rr];
    const float delta = (d > 20.f) ? d : log1pf(__expf(d));
    const float z  = xz[DINNER_ + c];
    const float sz = z / (1.f + __expf(-z));
    const float y  = (delta * s + Dskip[c]) * u_sh[wave][c] * sz;
    Y[(size_t)b * DINNER_ + c] = (_Float16)y;
  }
}

// ---------------------------------------------------------------------------
extern "C" void kernel_launch(void* const* d_in, const int* in_sizes, int n_in,
                              void* d_out, int out_size, void* d_ws, size_t ws_size,
                              hipStream_t stream) {
  const float* perception = (const float*)d_in[0];
  const float* W_in       = (const float*)d_in[1];
  const float* b_in       = (const float*)d_in[2];
  const float* mu_w       = (const float*)d_in[3];
  const float* mu_b       = (const float*)d_in[4];
  const float* ls_w       = (const float*)d_in[5];
  const float* ls_b       = (const float*)d_in[6];
  const float* in_proj_w  = (const float*)d_in[7];
  const float* conv_w     = (const float*)d_in[8];
  const float* conv_b     = (const float*)d_in[9];
  const float* x_proj_w   = (const float*)d_in[10];
  const float* dt_proj_w  = (const float*)d_in[11];
  const float* dt_proj_b  = (const float*)d_in[12];
  /* d_in[13] = A_log: provably unused (h0 == 0, L == 1) */
  const float* Dskip      = (const float*)d_in[14];
  const float* out_proj_w = (const float*)d_in[15];

  char* ws = (char*)d_ws;
  size_t off = 0;
  auto alloc = [&](size_t bytes) {
    size_t cur = off;
    off += (bytes + 255) & ~(size_t)255;
    return cur;
  };
  _Float16* pb       = (_Float16*)(ws + alloc((size_t)BATCH_ * P_ * 2));
  _Float16* w_in_h   = (_Float16*)(ws + alloc((size_t)DMODEL_ * P_ * 2));
  _Float16* inproj_h = (_Float16*)(ws + alloc((size_t)(2 * DINNER_) * DMODEL_ * 2));
  _Float16* outproj_h= (_Float16*)(ws + alloc((size_t)DMODEL_ * DINNER_ * 2));
  _Float16* muls_h   = (_Float16*)(ws + alloc((size_t)128 * DMODEL_ * 2));
  _Float16* X_h      = (_Float16*)(ws + alloc((size_t)BATCH_ * DMODEL_ * 2));
  float*    XZf      = (float*)   (ws + alloc((size_t)BATCH_ * 2 * DINNER_ * 4));
  _Float16* Y_h      = (_Float16*)(ws + alloc((size_t)BATCH_ * DINNER_ * 2));
  _Float16* H_h      = (_Float16*)(ws + alloc((size_t)BATCH_ * DMODEL_ * 2));
  (void)ws_size; (void)in_sizes; (void)n_in; (void)out_size;

  auto cvt = [&](const float* s, _Float16* d, int n) {
    int n4 = n / 4;
    int blocks = (n4 + 255) / 256;
    if (blocks > 2048) blocks = 2048;
    cvt_f32_f16<<<blocks, 256, 0, stream>>>(s, d, n4);
  };
  cvt(perception, pb,        BATCH_ * P_);
  cvt(W_in,       w_in_h,    DMODEL_ * P_);
  cvt(in_proj_w,  inproj_h,  2 * DINNER_ * DMODEL_);
  cvt(out_proj_w, outproj_h, DMODEL_ * DINNER_);
  cvt(mu_w,       muls_h,            64 * DMODEL_);
  cvt(ls_w,       muls_h + 64 * DMODEL_, 64 * DMODEL_);

  // GEMM1: X = perception @ W_in^T + b_in   -> f16
  gemm_wmma<0><<<dim3(DMODEL_ / 128, BATCH_ / 64), 256, 0, stream>>>(
      pb, w_in_h, X_h, b_in, nullptr, BATCH_, DMODEL_, P_);

  // GEMM2: XZ = X @ in_proj_w^T   -> f32
  gemm_wmma<1><<<dim3((2 * DINNER_) / 128, BATCH_ / 64), 256, 0, stream>>>(
      X_h, inproj_h, XZf, nullptr, nullptr, BATCH_, 2 * DINNER_, DMODEL_);

  // Middle: collapsed Mamba scan -> Y (f16)
  mamba_mid<<<BATCH_ / 8, 256, 0, stream>>>(
      XZf, conv_w, conv_b, x_proj_w, dt_proj_w, dt_proj_b, Dskip, Y_h);

  // GEMM3: H = Y @ out_proj_w^T   -> f16
  gemm_wmma<0><<<dim3(DMODEL_ / 128, BATCH_ / 64), 256, 0, stream>>>(
      Y_h, outproj_h, H_h, nullptr, nullptr, BATCH_, DMODEL_, DINNER_);

  // GEMM4: [mu|log_std] = H @ [mu_w;ls_w]^T, tanh/clip epilogue -> d_out
  gemm_wmma<2><<<dim3(128 / 128, BATCH_ / 64), 256, 0, stream>>>(
      H_h, muls_h, d_out, mu_b, ls_b, BATCH_, 128, DMODEL_);
}
